// SwinTCore_54932631716155
// MI455X (gfx1250) — matrix operand order repaired and checked by reading, not compiled, not run on hardware
//
#include <hip/hip_runtime.h>

typedef _Float16 v16h __attribute__((ext_vector_type(16)));
typedef _Float16 h8   __attribute__((ext_vector_type(8)));
typedef float    v8f  __attribute__((ext_vector_type(8)));

static __device__ __forceinline__ v8f wmma16(v16h a, v16h b, v8f c) {
  return __builtin_amdgcn_wmma_f32_16x16x32_f16(false, a, false, b, (short)0, c, false, false);
}

// A fragment (16x32 f16): per lane halves [khA..khA+15] contiguous
static __device__ __forceinline__ v16h load_a_frag(const _Float16* __restrict__ base, int khA) {
  h8 lo = *(const h8*)(base + khA);
  h8 hi = *(const h8*)(base + 8 + khA);
  return __builtin_shufflevector(lo, hi, 0,1,2,3,4,5,6,7,8,9,10,11,12,13,14,15);
}

// B fragment (32x16 f16): per lane halves [khB .. khB+15] contiguous
static __device__ __forceinline__ v16h load_b_frag(const _Float16* __restrict__ base, int khB) {
  return *(const v16h*)(base + khB);
}

// ---------------------------------------------------------------------------
// Generic GEMM: out[M,N] = A[M,K] @ W[N,K]^T (+bias f32) (gelu?) (+res f32)
// A, W f16; f32 accumulate; OutT in {float, _Float16}. One wave = 32x32 tile.
// ---------------------------------------------------------------------------
template <typename OutT>
__global__ __launch_bounds__(32) void gemm_wmma_kernel(
    const _Float16* __restrict__ A, const _Float16* __restrict__ W,
    const float* __restrict__ bias, const float* __restrict__ res,
    OutT* __restrict__ out, int M, int N, int K, int gelu)
{
  const int lane = threadIdx.x & 31;
  const int m0 = blockIdx.x * 32;
  const int n0 = blockIdx.y * 32;
  const int khA = (lane >> 4) * 16;
  const int khB = (lane >> 4) * 16;

  int r0 = m0 + (lane & 15);      if (r0 >= M) r0 = M - 1;
  int r1 = m0 + 16 + (lane & 15); if (r1 >= M) r1 = M - 1;
  const int c0 = n0 + (lane & 15);
  const int c1 = c0 + 16;

  const _Float16* __restrict__ a0p = A + (long)r0 * K;
  const _Float16* __restrict__ a1p = A + (long)r1 * K;
  const _Float16* __restrict__ w0p = W + (long)c0 * K;
  const _Float16* __restrict__ w1p = W + (long)c1 * K;

  v8f acc00 = {}, acc01 = {}, acc10 = {}, acc11 = {};
  for (int k0 = 0; k0 < K; k0 += 32) {
    v16h a0 = load_a_frag(a0p + k0, khA);
    v16h a1 = load_a_frag(a1p + k0, khA);
    v16h b0 = load_b_frag(w0p + k0, khB);
    v16h b1 = load_b_frag(w1p + k0, khB);
    acc00 = wmma16(a0, b0, acc00);
    acc01 = wmma16(a0, b1, acc01);
    acc10 = wmma16(a1, b0, acc10);
    acc11 = wmma16(a1, b1, acc11);
  }

  auto epi = [&](v8f acc, int mbase, int col) {
    float bv = bias ? bias[col] : 0.f;
#pragma unroll
    for (int r = 0; r < 8; ++r) {
      int mm = mbase + (lane >> 4) * 8 + r;
      if (mm < M) {
        float v = acc[r] + bv;
        if (gelu) v = 0.5f * v * (1.0f + erff(v * 0.70710678118f));
        if (res) v += res[(long)mm * N + col];
        out[(long)mm * N + col] = (OutT)v;
      }
    }
  };
  epi(acc00, m0, c0);
  epi(acc01, m0, c1);
  epi(acc10, m0 + 16, c0);
  epi(acc11, m0 + 16, c1);
}

// ---------------------------------------------------------------------------
// Per-layer bias table expansion: btab[head][64][64] = rel bias or -1e30 pad
// ---------------------------------------------------------------------------
__global__ __launch_bounds__(256) void bias_table_kernel(
    const float* __restrict__ rel, float* __restrict__ btab, int heads)
{
  int idx = blockIdx.x * 256 + threadIdx.x;
  if (idx >= heads * 4096) return;
  int head = idx >> 12;
  int ij = idx & 4095;
  int i = ij >> 6, j = ij & 63;
  float v = -1e30f;
  if (i < 49 && j < 49) {
    int ri = i / 7 - j / 7 + 6;
    int rj = i % 7 - j % 7 + 6;
    v = rel[(ri * 13 + rj) * heads + head];
  }
  btab[idx] = v;
}

// ---------------------------------------------------------------------------
// Fused window attention: one wave per (window, head). hd == 32 always.
// qkv f16 [token][3C]; out f16 [token][C]; scale folded into epilogue.
// ---------------------------------------------------------------------------
__global__ __launch_bounds__(32) void attn_kernel(
    const _Float16* __restrict__ qkv, const float* __restrict__ btab,
    _Float16* __restrict__ xattn,
    int Hc, int Wc, int C, int shift)
{
  const int lane = threadIdx.x & 31;
  const int win  = blockIdx.x;
  const int head = blockIdx.y;
  const int nH = Hc / 7, nW = Wc / 7;
  const int b  = win / (nH * nW);
  const int wr = (win / nW) % nH;
  const int wc = win % nW;
  const int C3 = 3 * C;
  const float scale = 0.1767766953f;   // 32^-0.5

  __shared__ float S[64][65];
  __shared__ __align__(16) _Float16 Pm[64][72];
  __shared__ __align__(16) _Float16 Vs[64][40];
  __shared__ int tt[64];
  __shared__ int regid[64];

  for (int i = lane; i < 64; i += 32) {
    int r = i / 7, c = i % 7;
    int h = wr * 7 + r, w = wc * 7 + c;
    int hs = (h + shift) % Hc, ws2 = (w + shift) % Wc;
    tt[i] = (i < 49) ? ((b * Hc + hs) * Wc + ws2) : 0;
    int rh = (h < Hc - 7) ? 0 : ((h < Hc - shift) ? 1 : 2);
    int rw = (w < Wc - 7) ? 0 : ((w < Wc - shift) ? 1 : 2);
    regid[i] = rh * 3 + rw;
  }
  __syncthreads();

  // stage V (64x32 f16) into LDS, 8-half vector copies through the token table
  for (int idx = lane; idx < 64 * 4; idx += 32) {
    int j = idx >> 2, d8 = (idx & 3) * 8;
    *(h8*)(&Vs[j][d8]) = *(const h8*)(qkv + (long)tt[j] * C3 + 2 * C + head * 32 + d8);
  }

  const int khA = (lane >> 4) * 16;
  const int khB = (lane >> 4) * 16;
  const float* __restrict__ bhead = btab + (long)head * 4096;

  // ----- S = scale*(q @ k^T) + bias (+mask), padded with -inf --------------
  for (int mt = 0; mt < 4; ++mt) {
    int row = mt * 16 + (lane & 15);
    v16h a = load_a_frag(qkv + (long)tt[row] * C3 + head * 32, khA);
    int rgi[8];
#pragma unroll
    for (int r8 = 0; r8 < 8; ++r8) rgi[r8] = regid[mt * 16 + (lane >> 4) * 8 + r8];
    for (int nt = 0; nt < 4; ++nt) {
      int colj = nt * 16 + (lane & 15);
      v16h bf = load_b_frag(qkv + (long)tt[colj] * C3 + C + head * 32, khB);
      v8f acc = {};
      acc = wmma16(a, bf, acc);
      int rgj = regid[colj];
      const float* __restrict__ brow = bhead + colj;
#pragma unroll
      for (int r8 = 0; r8 < 8; ++r8) {
        int iM = mt * 16 + (lane >> 4) * 8 + r8;
        float v2 = acc[r8] * scale + brow[iM * 64];
        if (shift > 0 && rgi[r8] != rgj) v2 -= 100.0f;
        S[iM][colj] = v2;
      }
    }
  }
  __syncthreads();

  // ----- row softmax -> f16 probabilities (single pass) --------------------
  for (int rr = lane; rr < 64; rr += 32) {
    float mx = -1e30f;
    for (int j = 0; j < 64; ++j) mx = fmaxf(mx, S[rr][j]);
    float sum = 0.f;
    float e[64];
    for (int j = 0; j < 64; ++j) { e[j] = __expf(S[rr][j] - mx); sum += e[j]; }
    float inv = 1.0f / sum;
    for (int j = 0; j < 64; ++j) Pm[rr][j] = (_Float16)(e[j] * inv);
  }
  __syncthreads();

  // ----- O = P @ V (V from LDS), scatter back in token order ---------------
  for (int mt = 0; mt < 4; ++mt) {
    int row = mt * 16 + (lane & 15);
    const _Float16* prow = &Pm[row][0];
    for (int nt = 0; nt < 2; ++nt) {
      int n = lane & 15;
      int d = nt * 16 + n;
      v8f acc = {};
#pragma unroll
      for (int kt = 0; kt < 2; ++kt) {
        h8 lo = *(const h8*)(prow + kt * 32 + khA);
        h8 hi = *(const h8*)(prow + kt * 32 + 8 + khA);
        v16h a = __builtin_shufflevector(lo, hi, 0,1,2,3,4,5,6,7,8,9,10,11,12,13,14,15);
        v16h bf;
#pragma unroll
        for (int v = 0; v < 8; ++v) {
          int j0 = kt * 32 + khB + 2 * v;
          bf[2*v]   = Vs[j0][d];
          bf[2*v+1] = Vs[j0 + 1][d];
        }
        acc = wmma16(a, bf, acc);
      }
#pragma unroll
      for (int r8 = 0; r8 < 8; ++r8) {
        int iM = mt * 16 + (lane >> 4) * 8 + r8;
        if (iM < 49)
          xattn[(long)tt[iM] * C + head * 32 + d] = (_Float16)acc[r8];
      }
    }
  }
}

// ---------------------------------------------------------------------------
// LayerNorm: one wave per token; OutT selects f32 (residual stream) or f16
// ---------------------------------------------------------------------------
template <typename OutT>
__global__ __launch_bounds__(256) void ln_kernel(
    const float* __restrict__ x, const float* __restrict__ g,
    const float* __restrict__ b, OutT* __restrict__ y, int M, int C)
{
  int wave = threadIdx.x >> 5, lane = threadIdx.x & 31;
  long t = (long)blockIdx.x * 8 + wave;
  if (t >= M) return;
  const float* xr = x + t * C;
  float s = 0.f;
  for (int c = lane; c < C; c += 32) s += xr[c];
  for (int m = 16; m; m >>= 1) s += __shfl_xor(s, m, 32);
  float mean = s / C;
  float vv = 0.f;
  for (int c = lane; c < C; c += 32) { float d = xr[c] - mean; vv += d * d; }
  for (int m = 16; m; m >>= 1) vv += __shfl_xor(vv, m, 32);
  float inv = rsqrtf(vv / C + 1e-5f);
  OutT* yr = y + t * C;
  for (int c = lane; c < C; c += 32) yr[c] = (OutT)((xr[c] - mean) * inv * g[c] + b[c]);
}

// ---------------------------------------------------------------------------
// Patch embed gather, zero-padded K=48 -> 64, output f16: (B*56*56, 64)
// ---------------------------------------------------------------------------
__global__ __launch_bounds__(256) void patch_gather_kernel(
    const float* __restrict__ xin, _Float16* __restrict__ xe, int B, int H, int W)
{
  long idx = (long)blockIdx.x * blockDim.x + threadIdx.x;
  long total = (long)B * H * W * 64;
  if (idx >= total) return;
  int f = (int)(idx & 63); long t = idx >> 6;
  float val = 0.f;
  if (f < 48) {
    int w = (int)(t % W); long t2 = t / W;
    int h = (int)(t2 % H); int b = (int)(t2 / H);
    int c = f / 16, i = (f / 4) % 4, j = f % 4;
    val = xin[(((long)b * 3 + c) * (H * 4) + (h * 4 + i)) * (long)(W * 4) + (w * 4 + j)];
  }
  xe[idx] = (_Float16)val;
}

// Pad patch weight (96,48) -> (96,64) f16
__global__ __launch_bounds__(256) void pad_patchw_kernel(
    const float* __restrict__ w, _Float16* __restrict__ wp)
{
  int i = blockIdx.x * 256 + threadIdx.x;
  if (i >= 96 * 64) return;
  int n = i >> 6, k = i & 63;
  wp[i] = (k < 48) ? (_Float16)w[n * 48 + k] : (_Float16)0.f;
}

// f32 -> f16 weight conversion
__global__ __launch_bounds__(256) void cvt_h_kernel(
    const float* __restrict__ src, _Float16* __restrict__ dst, long n)
{
  long i = (long)blockIdx.x * 256 + threadIdx.x;
  if (i < n) dst[i] = (_Float16)src[i];
}

// ---------------------------------------------------------------------------
// Patch merge gather + LayerNorm(4C) -> f16: one wave per output token
// ---------------------------------------------------------------------------
__global__ __launch_bounds__(256) void merge_ln_kernel(
    const float* __restrict__ x, const float* __restrict__ g,
    const float* __restrict__ bb, _Float16* __restrict__ y,
    int B, int H, int W, int C)
{
  int H2 = H / 2, W2 = W / 2, C4 = 4 * C;
  int wave = threadIdx.x >> 5, lane = threadIdx.x & 31;
  long t = (long)blockIdx.x * 8 + wave;
  if (t >= (long)B * H2 * W2) return;
  int w2 = (int)(t % W2); long t2 = t / W2;
  int h2 = (int)(t2 % H2); int b = (int)(t2 / H2);

  auto get = [&](int jj) -> float {
    int q = jj / C, cc = jj % C;
    int dh = q & 1, dw = q >> 1;
    return x[(((long)b * H + (2 * h2 + dh)) * W + (2 * w2 + dw)) * (long)C + cc];
  };
  float s = 0.f;
  for (int j = lane; j < C4; j += 32) s += get(j);
  for (int m = 16; m; m >>= 1) s += __shfl_xor(s, m, 32);
  float mean = s / C4;
  float vv = 0.f;
  for (int j = lane; j < C4; j += 32) { float d = get(j) - mean; vv += d * d; }
  for (int m = 16; m; m >>= 1) vv += __shfl_xor(vv, m, 32);
  float inv = rsqrtf(vv / C4 + 1e-5f);
  _Float16* yr = y + t * C4;
  for (int j = lane; j < C4; j += 32) yr[j] = (_Float16)((get(j) - mean) * inv * g[j] + bb[j]);
}

// ---------------------------------------------------------------------------
// Final LayerNorm + (B,H,W,C) -> (B,C,H,W) transpose into d_out (f32)
// ---------------------------------------------------------------------------
__global__ __launch_bounds__(256) void ln_out_kernel(
    const float* __restrict__ x, const float* __restrict__ g,
    const float* __restrict__ b, float* __restrict__ out,
    int B, int HW, int C)
{
  int wave = threadIdx.x >> 5, lane = threadIdx.x & 31;
  long t = (long)blockIdx.x * 8 + wave;
  if (t >= (long)B * HW) return;
  int p = (int)(t % HW); int bb = (int)(t / HW);
  const float* xr = x + t * C;
  float s = 0.f;
  for (int c = lane; c < C; c += 32) s += xr[c];
  for (int m = 16; m; m >>= 1) s += __shfl_xor(s, m, 32);
  float mean = s / C;
  float vv = 0.f;
  for (int c = lane; c < C; c += 32) { float d = xr[c] - mean; vv += d * d; }
  for (int m = 16; m; m >>= 1) vv += __shfl_xor(vv, m, 32);
  float inv = rsqrtf(vv / C + 1e-5f);
  for (int c = lane; c < C; c += 32)
    out[((long)bb * C + c) * HW + p] = (xr[c] - mean) * inv * g[c] + b[c];
}

// ---------------------------------------------------------------------------
// Host orchestration
// ---------------------------------------------------------------------------
static inline int cdiv(int a, int b) { return (a + b - 1) / b; }

template <typename OutT>
static inline void run_gemm(const _Float16* A, const _Float16* W, const float* bias,
                            const float* res, OutT* out, int M, int N, int K,
                            int gelu, hipStream_t s)
{
  dim3 grid(cdiv(M, 32), N / 32);
  gemm_wmma_kernel<OutT><<<grid, 32, 0, s>>>(A, W, bias, res, out, M, N, K, gelu);
}

extern "C" void kernel_launch(void* const* d_in, const int* in_sizes, int n_in,
                              void* d_out, int out_size, void* d_ws, size_t ws_size,
                              hipStream_t stream)
{
  (void)in_sizes; (void)n_in; (void)out_size; (void)ws_size;
  const int DEPTHS[4] = {2, 2, 6, 2};
  const int HEADS_[4] = {3, 6, 12, 24};
  const int CS[4]     = {96, 192, 384, 768};
  const int Bn = 16;

  int pi = 0;
  auto in = [&](int i) { return (const float*)d_in[i]; };

  const float* Xin      = in(pi++);
  const float* patch_w  = in(pi++);
  const float* patch_b  = in(pi++);
  const float* patch_ng = in(pi++);
  const float* patch_nb = in(pi++);

  struct Blk { const float *n1g,*n1b,*qkvw,*qkvb,*rel,*projw,*projb,*n2g,*n2b,*fc1w,*fc1b,*fc2w,*fc2b; };
  Blk blks[12];
  for (int k = 0; k < 12; ++k) {
    blks[k].n1g  = in(pi++); blks[k].n1b  = in(pi++);
    blks[k].qkvw = in(pi++); blks[k].qkvb = in(pi++);
    blks[k].rel  = in(pi++);
    blks[k].projw= in(pi++); blks[k].projb= in(pi++);
    blks[k].n2g  = in(pi++); blks[k].n2b  = in(pi++);
    blks[k].fc1w = in(pi++); blks[k].fc1b = in(pi++);
    blks[k].fc2w = in(pi++); blks[k].fc2b = in(pi++);
  }
  struct Mrg { const float *ng,*nb,*rw; } mrgs[3];
  for (int k = 0; k < 3; ++k) { mrgs[k].ng = in(pi++); mrgs[k].nb = in(pi++); mrgs[k].rw = in(pi++); }
  const float* norm_g = in(pi++);
  const float* norm_b = in(pi++);

  // workspace layout
  const long ACT  = 50176L * 96;        // residual-stream elements (f32)
  const long QKV  = 50176L * 288;
  const long HBUF = 50176L * 384;
  float* ws      = (float*)d_ws;
  float* actA    = ws;
  float* actB    = actA + ACT;
  _Float16* lnb   = (_Float16*)(actB + ACT);     // capacity 2*ACT halves
  _Float16* qkvbf = (_Float16*)(actB + 2 * ACT); // capacity 2*QKV halves
  _Float16* attb  = (_Float16*)((float*)qkvbf + QKV);
  _Float16* hb    = (_Float16*)((float*)attb + ACT);
  _Float16* wpool = (_Float16*)((float*)hb + HBUF);

  // ---- convert all weights to f16 once ----
  _Float16* wq[12]; _Float16* wp[12]; _Float16* w1[12]; _Float16* w2[12];
  _Float16* wr[3];  _Float16* wpatch;
  long wcur = 0;
  {
    auto cvt = [&](const float* src, long n) {
      _Float16* dst = wpool + wcur; wcur += n;
      cvt_h_kernel<<<(int)cdiv((int)n, 256), 256, 0, stream>>>(src, dst, n);
      return dst;
    };
    int bi = 0;
    for (int s = 0; s < 4; ++s) {
      long C = CS[s];
      for (int i = 0; i < DEPTHS[s]; ++i, ++bi) {
        wq[bi] = cvt(blks[bi].qkvw, 3 * C * C);
        wp[bi] = cvt(blks[bi].projw, C * C);
        w1[bi] = cvt(blks[bi].fc1w, 4 * C * C);
        w2[bi] = cvt(blks[bi].fc2w, 4 * C * C);
      }
      if (s < 3) wr[s] = cvt(mrgs[s].rw, 2 * C * (4 * C));
    }
    wpatch = wpool + wcur; wcur += 96 * 64;
    pad_patchw_kernel<<<cdiv(96 * 64, 256), 256, 0, stream>>>(patch_w, wpatch);
  }
  float* btab = (float*)(wpool + ((wcur + 3) & ~3L));  // heads*64*64 f32 per layer

  // ---- patch embed (K padded 48 -> 64) ----
  {
    long tot = (long)Bn * 56 * 56 * 64;
    patch_gather_kernel<<<(int)cdiv((int)tot, 256), 256, 0, stream>>>(Xin, hb, Bn, 56, 56);
    run_gemm<float>(hb, wpatch, patch_b, nullptr, actB, Bn * 56 * 56, 96, 64, 0, stream);
    ln_kernel<float><<<cdiv(Bn * 56 * 56, 8), 256, 0, stream>>>(
        actB, patch_ng, patch_nb, actA, Bn * 56 * 56, 96);
  }

  float* cur = actA;
  float* alt = actB;
  int Hc = 56, Wc = 56, C = 96;
  int bi = 0;

  for (int s = 0; s < 4; ++s) {
    int T = Bn * Hc * Wc;
    int heads = HEADS_[s];
    for (int i = 0; i < DEPTHS[s]; ++i, ++bi) {
      const Blk& p = blks[bi];
      int shift = ((i & 1) && Hc > 7) ? 3 : 0;

      ln_kernel<_Float16><<<cdiv(T, 8), 256, 0, stream>>>(cur, p.n1g, p.n1b, lnb, T, C);
      run_gemm<_Float16>(lnb, wq[bi], p.qkvb, nullptr, qkvbf, T, 3 * C, C, 0, stream);

      bias_table_kernel<<<cdiv(heads * 4096, 256), 256, 0, stream>>>(p.rel, btab, heads);
      dim3 agrid(Bn * (Hc / 7) * (Wc / 7), heads);
      attn_kernel<<<agrid, 32, 0, stream>>>(qkvbf, btab, attb, Hc, Wc, C, shift);

      run_gemm<float>(attb, wp[bi], p.projb, cur, cur, T, C, C, 0, stream);
      ln_kernel<_Float16><<<cdiv(T, 8), 256, 0, stream>>>(cur, p.n2g, p.n2b, lnb, T, C);
      run_gemm<_Float16>(lnb, w1[bi], p.fc1b, nullptr, hb, T, 4 * C, C, 1, stream);
      run_gemm<float>(hb, w2[bi], p.fc2b, cur, cur, T, C, 4 * C, 0, stream);
    }
    if (s < 3) {
      int T2 = T / 4;
      merge_ln_kernel<<<cdiv(T2, 8), 256, 0, stream>>>(cur, mrgs[s].ng, mrgs[s].nb, hb, Bn, Hc, Wc, C);
      run_gemm<float>(hb, wr[s], nullptr, nullptr, alt, T2, 2 * C, 4 * C, 0, stream);
      float* tmp = cur; cur = alt; alt = tmp;
      Hc /= 2; Wc /= 2; C *= 2;
    }
  }

  ln_out_kernel<<<cdiv(Bn * Hc * Wc, 8), 256, 0, stream>>>(
      cur, norm_g, norm_b, (float*)d_out, Bn, Hc * Wc, C);
}
